// ExtendedSympNet_44427141710160
// MI455X (gfx1250) — matrix operand order repaired and checked by reading, not compiled, not run on hardware
//
#include <hip/hip_runtime.h>
#include <cmath>

#define B_N      32768
#define LAT      64
#define HID      512
#define ACTIVE   4
#define MROWS    32          // batch rows per workgroup (2 WMMA M-tiles)
#define THREADS  256         // 8 wave32s
#define DSZ      (MROWS*HID) // halves per LDS activation/deriv buffer

typedef __attribute__((ext_vector_type(16))) __bf16 v16bf;
typedef __attribute__((ext_vector_type(8)))  __bf16 v8bf;
typedef __attribute__((ext_vector_type(8)))  float  v8f;

// Fast tanh: prefer native v_tanh_f32 (CDNA5 TRANS op), else branch-free
// exp2-based identity on the native v_exp_f32 pipe, else libm.
__device__ __forceinline__ float fast_tanh(float x) {
#if __has_builtin(__builtin_amdgcn_tanhf)
  return __builtin_amdgcn_tanhf(x);
#elif __has_builtin(__builtin_amdgcn_exp2f) && __has_builtin(__builtin_amdgcn_rcpf)
  // tanh(x) = 1 - 2/(exp(2x)+1); exp(2x) = exp2(2*log2(e)*x)
  float e = __builtin_amdgcn_exp2f(x * 2.88539008177793f);
  return 1.0f - 2.0f * __builtin_amdgcn_rcpf(e + 1.0f);
#else
  return tanhf(x);
#endif
}

// ---------------------------------------------------------------------------
// Weight packing into the WMMA B-tile lane layout (v_wmma_f32_16x16x32_bf16).
// B is K x N. Per 32x16 tile:
//   lane l <16 : column N=l,    halves i=0..15 hold K = kt*32 + i
//   lane l>=16 : column N=l-16, halves i=0..15 hold K = kt*32 + 16 + i
// flat packed index: ((ntile*KT + kt)*32 + lane)*16 + i
// ---------------------------------------------------------------------------
__global__ void pack_kernel(const float* __restrict__ src, __bf16* __restrict__ dst,
                            int K, int N, int transposed, int srcStride) {
  int idx = blockIdx.x * blockDim.x + threadIdx.x;
  if (idx >= K * N) return;
  int i     = idx & 15;
  int lane  = (idx >> 4) & 31;
  int rest  = idx >> 9;
  int KT    = K >> 5;
  int kt    = rest % KT;
  int ntile = rest / KT;
  int n = ntile * 16 + (lane & 15);
  int k = kt * 32 + ((lane >= 16) ? 16 : 0) + i;
  float v = transposed ? src[(size_t)n * srcStride + k]
                       : src[(size_t)k * srcStride + n];
  dst[idx] = (__bf16)v;
}

// A-tile (16x32 bf16) load from LDS, in the same contiguous k order as the B packing:
//   lane<16 : M=lane,    K = kt*32 + [0..15]
//   lane>=16: M=lane-16, K = kt*32 + [16..31]
__device__ __forceinline__ v16bf load_a_tile(const __bf16* s, int strideH, int kt, int lane) {
  int row  = lane & 15;
  int koff = (lane >> 4) << 4;                 // 0 or 16 halves
  const __bf16* p = s + row * strideH + kt * 32 + koff;
  union { v16bf v; v8bf h[2]; } u;
  u.h[0] = *(const v8bf*)p;                    // 16B -> ds_load_b128
  u.h[1] = *(const v8bf*)(p + 8);              // +16B
  return u.v;
}

// Fused GEMM layer over a 32 x (KT*32) input tile in LDS.
// FWD:  out = tanh(in @ W + b), deriv = 1 - out^2   (both stored bf16 in LDS)
// !FWD: out = (in @ Wt) * dMul                      (backward chain step)
template<int KT, bool FWD>
__device__ __forceinline__ void layer_gemm(const __bf16* __restrict__ inS, int inStrideH,
                                           const __bf16* __restrict__ Wp,
                                           const float*  __restrict__ bias,
                                           const __bf16* __restrict__ dMul,
                                           __bf16* __restrict__ outS,
                                           __bf16* __restrict__ dOut,
                                           int lane, int wave) {
  v8f acc[2][4] = {};
  for (int kt = 0; kt < KT; ++kt) {
    v16bf a0 = load_a_tile(inS,                  inStrideH, kt, lane);
    v16bf a1 = load_a_tile(inS + 16 * inStrideH, inStrideH, kt, lane);
    #pragma unroll
    for (int nt = 0; nt < 4; ++nt) {
      int ntile = wave * 4 + nt;
      v16bf b = ((const v16bf*)Wp)[(ntile * KT + kt) * 32 + lane];
      acc[0][nt] = __builtin_amdgcn_wmma_f32_16x16x32_bf16(false, a0, false, b, (short)0, acc[0][nt], false, false);
      acc[1][nt] = __builtin_amdgcn_wmma_f32_16x16x32_bf16(false, a1, false, b, (short)0, acc[1][nt], false, false);
    }
  }
  // epilogue: C layout -> row = mt*16 + v + 8*(lane>=16), col = nbase + (lane&15)
  #pragma unroll
  for (int nt = 0; nt < 4; ++nt) {
    int col = wave * 64 + nt * 16 + (lane & 15);
    float bv = FWD ? bias[col] : 0.0f;
    #pragma unroll
    for (int mt = 0; mt < 2; ++mt) {
      #pragma unroll
      for (int v = 0; v < 8; ++v) {
        int r = mt * 16 + v + ((lane >> 4) << 3);
        float x = acc[mt][nt][v];
        if (FWD) {
          float t = fast_tanh(x + bv);
          outS[r * HID + col] = (__bf16)t;
          dOut[r * HID + col] = (__bf16)(1.0f - t * t);
        } else {
          float dd = (float)dMul[r * HID + col];
          outS[r * HID + col] = (__bf16)(x * dd);
        }
      }
    }
  }
}

// ---------------------------------------------------------------------------
// Fused gradH: forward (5 layers) + backward to the first 4 z-columns.
// One workgroup = 32 batch rows, 8 waves each owning 64 output columns.
// ---------------------------------------------------------------------------
__global__ void __launch_bounds__(THREADS)
grad_kernel(const float* __restrict__ zg,
            const __bf16* Wf1, const __bf16* Wf2, const __bf16* Wf3, const __bf16* Wf4, const __bf16* Wf5,
            const __bf16* Wt1, const __bf16* Wt2, const __bf16* Wt3, const __bf16* Wt4, const __bf16* Wt5,
            const float* b1, const float* b2, const float* b3, const float* b4, const float* b5,
            const float* W6, float* __restrict__ gout) {
  extern __shared__ char smem[];
  __bf16* zS   = (__bf16*)(smem);                              // 32x64   bf16 (4 KB)
  __bf16* hA   = (__bf16*)(smem + 4096);                       // 32x512  bf16 (32 KB)
  __bf16* hB   = (__bf16*)(smem + 4096 + 32768);               // 32x512  bf16 (32 KB)
  __bf16* dS   = (__bf16*)(smem + 4096 + 65536);               // 5x 32x512 bf16 (160 KB)
  float*  accS = (float*) (smem + 4096 + 65536 + 163840);      // 32x16 f32 (2 KB)

  const int tid  = threadIdx.x;
  const int lane = tid & 31;
  const int wave = tid >> 5;
  const int rbase = blockIdx.x * MROWS;

  // stage z tile (32x64) into LDS as bf16; zero the reduction buffer
  for (int idx = tid; idx < MROWS * LAT; idx += THREADS) {
    int r = idx >> 6, c = idx & 63;
    zS[idx] = (__bf16)zg[(size_t)(rbase + r) * LAT + c];
  }
  for (int idx = tid; idx < MROWS * 16; idx += THREADS) accS[idx] = 0.0f;
  __syncthreads();

  // ---- forward ----
  layer_gemm< 2, true>(zS, LAT, Wf1, b1, nullptr, hA, dS + 0 * DSZ, lane, wave); __syncthreads();
  layer_gemm<16, true>(hA, HID, Wf2, b2, nullptr, hB, dS + 1 * DSZ, lane, wave); __syncthreads();
  layer_gemm<16, true>(hB, HID, Wf3, b3, nullptr, hA, dS + 2 * DSZ, lane, wave); __syncthreads();
  layer_gemm<16, true>(hA, HID, Wf4, b4, nullptr, hB, dS + 3 * DSZ, lane, wave); __syncthreads();
  layer_gemm<16, true>(hB, HID, Wf5, b5, nullptr, hA, dS + 4 * DSZ, lane, wave); __syncthreads();

  // ---- backward seed: u5 = W6 * (1 - h5^2) ----
  for (int idx = tid; idx < MROWS * HID; idx += THREADS) {
    int c = idx & (HID - 1);
    hB[idx] = (__bf16)(W6[c] * (float)dS[4 * DSZ + idx]);
  }
  __syncthreads();

  // ---- backward chain: u_{k-1} = (u_k @ W_k^T) * d_{k-1} ----
  layer_gemm<16, false>(hB, HID, Wt5, nullptr, dS + 3 * DSZ, hA, nullptr, lane, wave); __syncthreads();
  layer_gemm<16, false>(hA, HID, Wt4, nullptr, dS + 2 * DSZ, hB, nullptr, lane, wave); __syncthreads();
  layer_gemm<16, false>(hB, HID, Wt3, nullptr, dS + 1 * DSZ, hA, nullptr, lane, wave); __syncthreads();
  layer_gemm<16, false>(hA, HID, Wt2, nullptr, dS + 0 * DSZ, hB, nullptr, lane, wave); __syncthreads();

  // ---- gz = u1 @ W1^T : only N-tile 0 (cols 0..15; cols 0..3 are used).
  // K=512 split over 8 waves (2 k-tiles each), reduced via LDS float atomics.
  {
    v8f acc[2] = {};
    #pragma unroll
    for (int kk = 0; kk < 2; ++kk) {
      int kt = wave * 2 + kk;
      v16bf b = ((const v16bf*)Wt1)[kt * 32 + lane];   // ntile 0
      #pragma unroll
      for (int mt = 0; mt < 2; ++mt) {
        v16bf a = load_a_tile(hB + mt * 16 * HID, HID, kt, lane);
        acc[mt] = __builtin_amdgcn_wmma_f32_16x16x32_bf16(false, a, false, b, (short)0, acc[mt], false, false);
      }
    }
    #pragma unroll
    for (int mt = 0; mt < 2; ++mt) {
      #pragma unroll
      for (int v = 0; v < 8; ++v) {
        int r = mt * 16 + v + ((lane >> 4) << 3);
        atomicAdd(&accS[r * 16 + (lane & 15)], acc[mt][v]);
      }
    }
    __syncthreads();
    if (tid < MROWS * ACTIVE) {
      int r = tid >> 2, c = tid & 3;
      gout[(size_t)(rbase + r) * ACTIVE + c] = accS[r * 16 + c];
    }
  }
}

// ---------------------------------------------------------------------------
// Verlet elementwise updates on z[:, :4]
// ---------------------------------------------------------------------------
__global__ void update_full(float* __restrict__ z, const float* __restrict__ g,
                            const float* __restrict__ S, const float* __restrict__ dtq,
                            const float* __restrict__ dtp, const float* __restrict__ alpha,
                            float dt) {
  int i = blockIdx.x * blockDim.x + threadIdx.x;
  if (i >= B_N) return;
  float za[4], gg[4];
  #pragma unroll
  for (int c = 0; c < 4; ++c) { za[c] = z[(size_t)i * LAT + c]; gg[c] = g[(size_t)i * 4 + c]; }
  float sS[4], sT[4];
  #pragma unroll
  for (int j = 0; j < 4; ++j) {
    sS[j] = za[0]*S[0*4+j] + za[1]*S[1*4+j] + za[2]*S[2*4+j] + za[3]*S[3*4+j]; // z @ S
    sT[j] = za[0]*S[j*4+0] + za[1]*S[j*4+1] + za[2]*S[j*4+2] + za[3]*S[j*4+3]; // z @ S^T
  }
  float al = *alpha, q = *dtq, p = *dtp;
  float z2n0 = za[2] + (-gg[0]*p + al*sS[2]) * (dt*0.5f);
  float z2n1 = za[3] + (-gg[1]*p + al*sS[3]) * (dt*0.5f);
  float z1n0 = za[0] + ( gg[2]*q + al*sT[0]) * dt;
  float z1n1 = za[1] + ( gg[3]*q + al*sT[1]) * dt;
  z[(size_t)i * LAT + 0] = z1n0;
  z[(size_t)i * LAT + 1] = z1n1;
  z[(size_t)i * LAT + 2] = z2n0;
  z[(size_t)i * LAT + 3] = z2n1;
}

__global__ void update_half(float* __restrict__ z, const float* __restrict__ g,
                            const float* __restrict__ S, const float* __restrict__ dtp,
                            const float* __restrict__ alpha, float dt) {
  int i = blockIdx.x * blockDim.x + threadIdx.x;
  if (i >= B_N) return;
  float za[4];
  #pragma unroll
  for (int c = 0; c < 4; ++c) za[c] = z[(size_t)i * LAT + c];
  float al = *alpha, p = *dtp;
  float sS2 = za[0]*S[0*4+2] + za[1]*S[1*4+2] + za[2]*S[2*4+2] + za[3]*S[3*4+2];
  float sS3 = za[0]*S[0*4+3] + za[1]*S[1*4+3] + za[2]*S[2*4+3] + za[3]*S[3*4+3];
  z[(size_t)i * LAT + 2] = za[2] + (-g[(size_t)i*4+0]*p + al*sS2) * (dt*0.5f);
  z[(size_t)i * LAT + 3] = za[3] + (-g[(size_t)i*4+1]*p + al*sS3) * (dt*0.5f);
}

// ---------------------------------------------------------------------------
extern "C" void kernel_launch(void* const* d_in, const int* in_sizes, int n_in,
                              void* d_out, int out_size, void* d_ws, size_t ws_size,
                              hipStream_t stream) {
  const float* z   = (const float*)d_in[0];
  const float* W1  = (const float*)d_in[1];
  const float* b1  = (const float*)d_in[2];
  const float* W2  = (const float*)d_in[3];
  const float* b2  = (const float*)d_in[4];
  const float* W3  = (const float*)d_in[5];
  const float* b3  = (const float*)d_in[6];
  const float* W4  = (const float*)d_in[7];
  const float* b4  = (const float*)d_in[8];
  const float* W5  = (const float*)d_in[9];
  const float* b5  = (const float*)d_in[10];
  const float* W6  = (const float*)d_in[11];
  const float* S   = (const float*)d_in[13];
  const float* dtq = (const float*)d_in[14];
  const float* dtp = (const float*)d_in[15];
  const float* alp = (const float*)d_in[16];

  char* ws = (char*)d_ws;
  constexpr size_t SZ_Z  = (size_t)B_N * LAT * 4;        // 8 MB
  constexpr size_t SZ_G  = (size_t)B_N * ACTIVE * 4;     // 512 KB
  constexpr size_t SZ_W1 = (size_t)LAT * HID * 2;        // 64 KB bf16
  constexpr size_t SZ_W  = (size_t)HID * HID * 2;        // 512 KB bf16
  size_t off = 0;
  float*  zcur = (float*)(ws + off); off += SZ_Z;
  float*  gbuf = (float*)(ws + off); off += SZ_G;
  __bf16* Wf1  = (__bf16*)(ws + off); off += SZ_W1;
  __bf16* Wf2  = (__bf16*)(ws + off); off += SZ_W;
  __bf16* Wf3  = (__bf16*)(ws + off); off += SZ_W;
  __bf16* Wf4  = (__bf16*)(ws + off); off += SZ_W;
  __bf16* Wf5  = (__bf16*)(ws + off); off += SZ_W;
  __bf16* Wt1  = (__bf16*)(ws + off); off += SZ_W1;
  __bf16* Wt2  = (__bf16*)(ws + off); off += SZ_W;
  __bf16* Wt3  = (__bf16*)(ws + off); off += SZ_W;
  __bf16* Wt4  = (__bf16*)(ws + off); off += SZ_W;
  __bf16* Wt5  = (__bf16*)(ws + off); off += SZ_W;
  (void)ws_size; (void)in_sizes; (void)n_in; (void)out_size;

  auto pk = [&](const float* src, __bf16* dst, int K, int N, int tr, int stride) {
    int total = K * N;
    pack_kernel<<<(total + 255) / 256, 256, 0, stream>>>(src, dst, K, N, tr, stride);
  };
  // forward-layout packs (B = W, K x N)
  pk(W1, Wf1,  64, 512, 0, 512);
  pk(W2, Wf2, 512, 512, 0, 512);
  pk(W3, Wf3, 512, 512, 0, 512);
  pk(W4, Wf4, 512, 512, 0, 512);
  pk(W5, Wf5, 512, 512, 0, 512);
  // transposed packs (B = W^T)
  pk(W1, Wt1, 512,  64, 1, 512);
  pk(W2, Wt2, 512, 512, 1, 512);
  pk(W3, Wt3, 512, 512, 1, 512);
  pk(W4, Wt4, 512, 512, 1, 512);
  pk(W5, Wt5, 512, 512, 1, 512);

  hipMemcpyAsync(zcur, z, SZ_Z, hipMemcpyDeviceToDevice, stream);

  constexpr size_t LDSB = 4096 + 2 * 32768 + 5 * 32768 + 2048; // 235,520 B
  const double a1 = 1.0 / (4.0 - pow(4.0, 1.0 / 3.0));
  const double a3 = 1.0 - 4.0 * a1;
  const double as[5] = {a1, a1, a3, a1, a1};

  for (int s = 0; s < 5; ++s) {
    float dt = (float)(as[s] * 0.1);
    grad_kernel<<<B_N / MROWS, THREADS, LDSB, stream>>>(zcur,
        Wf1, Wf2, Wf3, Wf4, Wf5, Wt1, Wt2, Wt3, Wt4, Wt5,
        b1, b2, b3, b4, b5, W6, gbuf);
    update_full<<<B_N / 256, 256, 0, stream>>>(zcur, gbuf, S, dtq, dtp, alp, dt);
    grad_kernel<<<B_N / MROWS, THREADS, LDSB, stream>>>(zcur,
        Wf1, Wf2, Wf3, Wf4, Wf5, Wt1, Wt2, Wt3, Wt4, Wt5,
        b1, b2, b3, b4, b5, W6, gbuf);
    update_half<<<B_N / 256, 256, 0, stream>>>(zcur, gbuf, S, dtp, alp, dt);
  }

  hipMemcpyAsync(d_out, zcur, SZ_Z, hipMemcpyDeviceToDevice, stream);
}
